// OrientedCoordinateAttention_26499948216458
// MI455X (gfx1250) — compile-verified, matrix operand order checked by reading, not run on hardware
//
#include <hip/hip_runtime.h>
#include <hip/hip_bf16.h>
#include <math.h>
#include <stdint.h>

#define Bn 8
#define Hh 256
#define Ww 256
#define Cc 64
// HIDDEN = 8, LENGTH = 9

// ---- kernel B tiling ----
#define TILE_W 16
#define TILE_H 16
#define HALO_LO 4
#define HALO_HI 5
#define TROWS (TILE_H + HALO_LO + HALO_HI)   // 25
#define TCOLS (TILE_W + HALO_LO + HALO_HI)   // 25
#define PP    68                             // pixel pitch in floats (64 + 4 pad; 272B: 16B aligned, 2-way bank conflict max)
#define ROWP  (TCOLS * PP)                   // 1700 floats
#define TILE_FLOATS (TROWS * ROWP)           // 42500 floats = 170000 B (dynamic LDS)

typedef __attribute__((ext_vector_type(16))) _Float16 v16h;
typedef __attribute__((ext_vector_type(8)))  float    v8f;

__device__ __forceinline__ int iclamp(int v, int lo, int hi) {
  return v < lo ? lo : (v > hi ? hi : v);
}
__device__ __forceinline__ float sigmoidf_(float z) {
  return 1.0f / (1.0f + __expf(-z));
}

// CDNA5 async global -> LDS copy (16 bytes), tracked by ASYNCcnt.
__device__ __forceinline__ void async_copy_b128(uint32_t lds_off, const void* gptr) {
  asm volatile("global_load_async_to_lds_b128 %0, %1, off"
               :: "v"(lds_off), "v"((unsigned long long)(uintptr_t)gptr)
               : "memory");
}
__device__ __forceinline__ void async_wait0() {
  asm volatile("s_wait_asynccnt 0x0" ::: "memory");
}

// =====================================================================
// Kernel A: conv3x3(64->8)+ReLU, conv1x1(8->1), sigmoid -> theta -> cos/sin
// One wave computes 16 consecutive pixels of one row via WMMA f16 (K=576).
// =====================================================================
__global__ __launch_bounds__(256) void theta_kernel(
    const float* __restrict__ x, const float* __restrict__ w3,
    const float* __restrict__ b3v, const float* __restrict__ w1,
    const float* __restrict__ b1v, float* __restrict__ cbuf,
    float* __restrict__ sbuf) {
  // Pre-swizzled B fragments for V_WMMA_F32_16X16X32_F16: [chunk 0..17][lane 0..31][16 halves]
  __shared__ __align__(32) _Float16 bfrag[18 * 32 * 16];
  __shared__ float red[8][16][16];

  const int tid = threadIdx.x;

  // Cooperative fill of w3 fragments (N padded 8 -> 16 with zeros).
  // 16-bit A/B fragment intra-lane K map: lane<16 -> K {0..7,16..23}; lane>=16 -> K {8..15,24..31}.
  for (int i = tid; i < 18 * 32 * 16; i += 256) {
    int chunk = i >> 9;
    int lane  = (i >> 4) & 31;
    int j     = i & 15;
    int n = lane & 15;
    int klocal = ((j < 8) ? j : (j + 8)) + ((lane & 16) ? 8 : 0);
    int k = chunk * 32 + klocal;          // K index in [0,576)
    int tap = k >> 6, c = k & 63;         // (ky*3+kx), channel
    float v = (n < 8) ? w3[(tap * 64 + c) * 8 + n] : 0.0f;
    bfrag[i] = (_Float16)v;
  }
  __syncthreads();

  const int wave = tid >> 5;
  const int lane = tid & 31;
  const int bid  = blockIdx.x;
  const int xblk = bid & 1;
  const int y    = (bid >> 1) & (Hh - 1);
  const int b    = bid >> 9;
  const int m    = lane & 15;                        // A-matrix row (pixel in wave)
  const int px   = xblk * 128 + wave * 16 + m;       // global x coordinate
  const int hiK  = (lane & 16) ? 8 : 0;

  v8f acc = {};
#pragma unroll
  for (int chunk = 0; chunk < 18; ++chunk) {
    const int tap = chunk >> 1;
    const int ky = tap / 3, kx = tap - ky * 3;
    const int c0 = ((chunk & 1) << 5) + hiK;         // channel base of this lane's low run
    const int yy = y + ky - 1;
    const int xx = px + kx - 1;
    // Branchless zero padding: clamp address, zero values via mask.
    const int inb = (yy >= 0) & (yy < Hh) & (xx >= 0) & (xx < Ww);
    const float msk = inb ? 1.0f : 0.0f;
    const int yyc = iclamp(yy, 0, Hh - 1);
    const int xxc = iclamp(xx, 0, Ww - 1);
    const float4* p =
        (const float4*)(x + ((((size_t)b * Hh + yyc) * Ww + xxc) << 6) + c0);
    float4 r0 = p[0], r1 = p[1];   // channels c0 .. c0+7
    float4 r2 = p[4], r3 = p[5];   // channels c0+16 .. c0+23
    v16h a;
    a[0]  = (_Float16)(r0.x * msk); a[1]  = (_Float16)(r0.y * msk);
    a[2]  = (_Float16)(r0.z * msk); a[3]  = (_Float16)(r0.w * msk);
    a[4]  = (_Float16)(r1.x * msk); a[5]  = (_Float16)(r1.y * msk);
    a[6]  = (_Float16)(r1.z * msk); a[7]  = (_Float16)(r1.w * msk);
    a[8]  = (_Float16)(r2.x * msk); a[9]  = (_Float16)(r2.y * msk);
    a[10] = (_Float16)(r2.z * msk); a[11] = (_Float16)(r2.w * msk);
    a[12] = (_Float16)(r3.x * msk); a[13] = (_Float16)(r3.y * msk);
    a[14] = (_Float16)(r3.z * msk); a[15] = (_Float16)(r3.w * msk);

    v16h bfv = *(const v16h*)&bfrag[(size_t)(((chunk << 5) + lane) << 4)];
    acc = __builtin_amdgcn_wmma_f32_16x16x32_f16(false, a, false, bfv,
                                                 (short)0, acc, false, false);
  }

  // Epilogue: bias + ReLU + 8->1 conv (per-pixel reduction via LDS).
  const int n = lane & 15;
  const float w1f = (n < 8) ? w1[n] : 0.0f;
  const float b3f = (n < 8) ? b3v[n] : 0.0f;
  const int moff = (lane & 16) ? 8 : 0;
#pragma unroll
  for (int rr = 0; rr < 8; ++rr) {
    float h = fmaxf(acc[rr] + b3f, 0.0f);
    red[wave][rr + moff][n] = h * w1f;
  }
  __syncthreads();
  if (lane < 16) {
    float s = b1v[0];
#pragma unroll
    for (int nn = 0; nn < 16; ++nn) s += red[wave][lane][nn];
    float theta = 3.14159265358979323f * sigmoidf_(s);
    float snt, cst;
    __sincosf(theta, &snt, &cst);
    const size_t pix = ((size_t)b * Hh + y) * Ww + xblk * 128 + wave * 16 + lane;
    cbuf[pix] = cst;
    sbuf[pix] = snt;
  }
}

// =====================================================================
// Kernel B: oriented bilinear pooling (from LDS tile) + 1x1 convs + gate.
// 256 threads = one 16x16 pixel tile; dynamic LDS tile with halo, staged
// via CDNA5 async global->LDS copies.
// =====================================================================
__device__ __forceinline__ void oriented_accum(const float* tile, float fy, float fx,
                                               float vy, float vx, int ty0, int tx0,
                                               int cc, float4 acc[4]) {
  for (int t = -4; t <= 4; ++t) {
    const float tf = (float)t;
    const float cy = fy + tf * vy;
    const float cx = fx + tf * vx;
    const float y0f = floorf(cy), x0f = floorf(cx);
    const int y0i = iclamp((int)y0f, 0, Hh - 1);
    const int x0i = iclamp((int)x0f, 0, Ww - 1);
    const int y1i = iclamp((int)(y0f + 1.0f), 0, Hh - 1);
    const int x1i = iclamp((int)(x0f + 1.0f), 0, Ww - 1);
    const float y0c = (float)y0i, x0c = (float)x0i;
    const float y1c = (float)y1i, x1c = (float)x1i;
    const float wa = (x1c - cx) * (y1c - cy);
    const float wb = (x1c - cx) * (cy - y0c);
    const float wc = (cx - x0c) * (y1c - cy);
    const float wd = (cx - x0c) * (cy - y0c);
    const int ry0 = iclamp(y0i - ty0 + HALO_LO, 0, TROWS - 1);
    const int ry1 = iclamp(y1i - ty0 + HALO_LO, 0, TROWS - 1);
    const int rx0 = iclamp(x0i - tx0 + HALO_LO, 0, TCOLS - 1);
    const int rx1 = iclamp(x1i - tx0 + HALO_LO, 0, TCOLS - 1);
    const float4* Pa = (const float4*)&tile[ry0 * ROWP + rx0 * PP + cc];
    const float4* Pb = (const float4*)&tile[ry1 * ROWP + rx0 * PP + cc];
    const float4* Pc = (const float4*)&tile[ry0 * ROWP + rx1 * PP + cc];
    const float4* Pd = (const float4*)&tile[ry1 * ROWP + rx1 * PP + cc];
#pragma unroll
    for (int q = 0; q < 4; ++q) {
      float4 A = Pa[q], B = Pb[q], C = Pc[q], D = Pd[q];
      acc[q].x += wa * A.x + wb * B.x + wc * C.x + wd * D.x;
      acc[q].y += wa * A.y + wb * B.y + wc * C.y + wd * D.y;
      acc[q].z += wa * A.z + wb * B.z + wc * C.z + wd * D.z;
      acc[q].w += wa * A.w + wb * B.w + wc * C.w + wd * D.w;
    }
  }
}

__global__ __launch_bounds__(256) void fuse_kernel(
    const float* __restrict__ x, const float* __restrict__ wr,
    const float* __restrict__ br, const float* __restrict__ we,
    const float* __restrict__ be, const float* __restrict__ cbuf,
    const float* __restrict__ sbuf, float* __restrict__ out) {
  extern __shared__ float tile[];          // TILE_FLOATS (170 KB)
  __shared__ float wrL[128 * 8];           // [ch][hidden], rows are 32B
  __shared__ float weT[128 * 8];           // transposed: [ch][hidden]
  __shared__ float brL[8];
  __shared__ float beL[128];

  const int tid = threadIdx.x;
  const int bid = blockIdx.x;
  const int bx = bid & 15;
  const int by = (bid >> 4) & 15;
  const int b  = bid >> 8;
  const int tx0 = bx << 4;
  const int ty0 = by << 4;

  // Stage the pixel tile (with clamped halo) into LDS via async copies.
  // Low 32 bits of a generic LDS pointer == byte offset in workgroup LDS.
  const uint32_t tile_base = (uint32_t)(uintptr_t)(const void*)tile;
  for (int slot = tid; slot < TROWS * TCOLS; slot += 256) {
    int trow = slot / TCOLS, tcol = slot - trow * TCOLS;
    int gy = iclamp(ty0 + trow - HALO_LO, 0, Hh - 1);
    int gx = iclamp(tx0 + tcol - HALO_LO, 0, Ww - 1);
    const float* src = x + ((((size_t)b * Hh + gy) * Ww + gx) << 6);
    uint32_t dst = tile_base + (uint32_t)((trow * ROWP + tcol * PP) * 4);
#pragma unroll
    for (int q = 0; q < 16; ++q) async_copy_b128(dst + q * 16, src + q * 4);
  }

  // Weight staging overlaps with the async tile DMA.
  for (int i = tid; i < 1024; i += 256) {
    wrL[i] = wr[i];
    int j = i >> 7, ch = i & 127;
    weT[ch * 8 + j] = we[i];
  }
  if (tid < 128) beL[tid] = be[tid];
  if (tid < 8)   brL[tid] = br[tid];

  async_wait0();
  __syncthreads();

  const int tx = tid & 15, ty = tid >> 4;
  const int gx = tx0 + tx, gy = ty0 + ty;
  const size_t pix = ((size_t)b * Hh + gy) * Ww + gx;
  const float ct = cbuf[pix], st = sbuf[pix];
  const float fy = (float)gy, fx = (float)gx;

  float r[8];
#pragma unroll
  for (int j = 0; j < 8; ++j) r[j] = brL[j];

  const float inv9 = 1.0f / 9.0f;
  for (int cc = 0; cc < 64; cc += 16) {
    float4 accT[4], accN[4];
#pragma unroll
    for (int q = 0; q < 4; ++q) {
      accT[q] = make_float4(0.f, 0.f, 0.f, 0.f);
      accN[q] = make_float4(0.f, 0.f, 0.f, 0.f);
    }
    // tangential: (vx,vy) = (cos, sin); normal: (vx,vy) = (-sin, cos)
    oriented_accum(tile, fy, fx, /*vy=*/st, /*vx=*/ct,  ty0, tx0, cc, accT);
    oriented_accum(tile, fy, fx, /*vy=*/ct, /*vx=*/-st, ty0, tx0, cc, accN);
#pragma unroll
    for (int q = 0; q < 4; ++q) {
      const float* at = (const float*)&accT[q];
      const float* an = (const float*)&accN[q];
#pragma unroll
      for (int k = 0; k < 4; ++k) {
        int ch = cc + q * 4 + k;
        float vt = at[k] * inv9;
        float vn = an[k] * inv9;
        const float* wt_ = &wrL[ch * 8];
        const float* wn_ = &wrL[(64 + ch) * 8];
#pragma unroll
        for (int j = 0; j < 8; ++j) r[j] += vt * wt_[j] + vn * wn_[j];
      }
    }
  }
#pragma unroll
  for (int j = 0; j < 8; ++j) r[j] = fmaxf(r[j], 0.0f);

  const float* myx = &tile[(ty + HALO_LO) * ROWP + (tx + HALO_LO) * PP];
  float* outp = out + (pix << 6);
  for (int c = 0; c < 64; c += 4) {
    float4 xv = *(const float4*)&myx[c];
    const float* xc = (const float*)&xv;
    float4 o;
    float* oc = (float*)&o;
#pragma unroll
    for (int k = 0; k < 4; ++k) {
      int ch = c + k;
      float zt = beL[ch], zn = beL[64 + ch];
      const float* wt_ = &weT[ch * 8];
      const float* wn_ = &weT[(64 + ch) * 8];
#pragma unroll
      for (int j = 0; j < 8; ++j) { zt += r[j] * wt_[j]; zn += r[j] * wn_[j]; }
      oc[k] = (sigmoidf_(zt) + sigmoidf_(zn)) * xc[k];
    }
    *(float4*)&outp[c] = o;
  }
}

extern "C" void kernel_launch(void* const* d_in, const int* in_sizes, int n_in,
                              void* d_out, int out_size, void* d_ws, size_t ws_size,
                              hipStream_t stream) {
  (void)in_sizes; (void)n_in; (void)out_size; (void)ws_size;
  const float* x  = (const float*)d_in[0];
  const float* w3 = (const float*)d_in[1];
  const float* b3 = (const float*)d_in[2];
  const float* w1 = (const float*)d_in[3];
  const float* b1 = (const float*)d_in[4];
  const float* wr = (const float*)d_in[5];
  const float* br = (const float*)d_in[6];
  const float* we = (const float*)d_in[7];
  const float* be = (const float*)d_in[8];

  float* cbuf = (float*)d_ws;                       // [B*H*W] cos(theta)
  float* sbuf = cbuf + (size_t)Bn * Hh * Ww;        // [B*H*W] sin(theta)

  // Kernel A: 8 waves/block, each wave = 16 pixels of one row.
  theta_kernel<<<Bn * Hh * (Ww / 128), 256, 0, stream>>>(x, w3, b3, w1, b1,
                                                         cbuf, sbuf);

  // Kernel B: one 16x16 tile per 256-thread block, big dynamic LDS tile.
  const size_t tileBytes = (size_t)TILE_FLOATS * sizeof(float);  // 170000 B
  hipFuncSetAttribute((const void*)fuse_kernel,
                      hipFuncAttributeMaxDynamicSharedMemorySize,
                      (int)tileBytes);
  fuse_kernel<<<Bn * (Hh / TILE_H) * (Ww / TILE_W), 256, tileBytes, stream>>>(
      x, wr, br, we, be, cbuf, sbuf, (float*)d_out);
}
